// DualVAE_5179730559176
// MI455X (gfx1250) — matrix operand order repaired and checked by reading, not compiled, not run on hardware
//
#include <hip/hip_runtime.h>
#include <cstdint>

// ---------------- problem constants ----------------
constexpr int Uc = 4096;   // users
constexpr int Ic = 8192;   // items
constexpr int Lc = 256;    // latent
constexpr int H1c = 512;
constexpr int H2c = 256;

typedef __attribute__((ext_vector_type(16))) _Float16 v16h;
typedef __attribute__((ext_vector_type(8)))  _Float16 v8h;
typedef __attribute__((ext_vector_type(8)))  float    v8f;

// gfx1250 async global->LDS copy (16B per lane). lds = wave-relative LDS byte
// address (low 32 bits of generic pointer), gaddr = 64-bit global address.
#define ASYNC_B128(lds, gaddr)                                                \
    asm volatile("global_load_async_to_lds_b128 %0, %1, off"                  \
                 :: "v"(lds), "v"(gaddr) : "memory")
#define WAIT_ASYNC6 asm volatile("s_wait_asynccnt 0x6" ::: "memory")
#define WAIT_ASYNC0 asm volatile("s_wait_asynccnt 0x0" ::: "memory")

// =====================================================================
// WMMA GEMM:  C[M,N] = epilogue( A[M,K] @ B^T + bias + scalar )
//   A:  f16 row-major [M,K]
//   Bt: f16 row-major [N,K]   (all weights pre-transposed; item_z is native)
//   bias: fp32[N] or null; sadd: fp32 scalar ptr or null
//   Cf: fp32 out (nullable); Ch: f16 out (nullable); act=1 -> tanh
// Tile: BM=128 BN=256 BK=32; 256 threads = 8 waves, each wave a 64x64 patch
// (4x4 accumulators -> 16 v_wmma per k-step per wave). Async double-buffered
// global->LDS staging, pipeline unrolled x2 so buffer indices are constants.
// Requires M%128==0, N%256==0, K%64==0 (true for all shapes here).
// =====================================================================
#define BMt 128
#define BNt 256
#define BKt 32

__global__ __launch_bounds__(256)
void gemm_f16(const _Float16* __restrict__ A, const _Float16* __restrict__ Bt,
              const float* __restrict__ bias, const float* __restrict__ sadd,
              float* __restrict__ Cf, _Float16* __restrict__ Ch,
              int M, int N, int K, int act)
{
    __shared__ alignas(32) _Float16 As[2][BMt * BKt];   // m-major [128][32]
    __shared__ alignas(32) _Float16 Bs[2][BNt * BKt];   // n-major [256][32]

    const int tid  = threadIdx.x;
    const int wid  = tid >> 5;
    const int lane = tid & 31;
    const int ml   = lane & 15;   // row (A) / col (B,C) within 16
    const int hi   = lane >> 4;   // K-half (frags) / M-half (C)

    const int m0 = blockIdx.y * BMt;
    const int n0 = blockIdx.x * BNt;
    const int wm = (wid & 1) * 64;    // 2 wave-rows of 64
    const int wn = (wid >> 1) * 64;   // 4 wave-cols of 64

    // ---- per-thread staging coordinates (64B-per-4-lanes coalescing) ----
    unsigned aLds[2][2], bLds[2][4];
    const _Float16* aG[2];
    const _Float16* bG[4];
#pragma unroll
    for (int j = 0; j < 2; ++j) {                     // A: 128 rows x 4 chunks
        int ci = j * 256 + tid, row = ci >> 2, c = ci & 3;
        aG[j] = A + (size_t)(m0 + row) * K + c * 8;
#pragma unroll
        for (int b = 0; b < 2; ++b)
            aLds[b][j] = (unsigned)(uintptr_t)&As[b][row * BKt + c * 8];
    }
#pragma unroll
    for (int j = 0; j < 4; ++j) {                     // B: 256 rows x 4 chunks
        int ci = j * 256 + tid, n = ci >> 2, c = ci & 3;
        bG[j] = Bt + (size_t)(n0 + n) * K + c * 8;
#pragma unroll
        for (int b = 0; b < 2; ++b)
            bLds[b][j] = (unsigned)(uintptr_t)&Bs[b][n * BKt + c * 8];
    }

    v8f acc[4][4] = {};

    auto issue = [&](int buf, int kOff) {             // buf is a literal at call sites
#pragma unroll
        for (int j = 0; j < 2; ++j)
            ASYNC_B128(aLds[buf][j], (unsigned long long)(uintptr_t)(aG[j] + kOff));
#pragma unroll
        for (int j = 0; j < 4; ++j)
            ASYNC_B128(bLds[buf][j], (unsigned long long)(uintptr_t)(bG[j] + kOff));
    };
    auto compute = [&](const _Float16* as, const _Float16* bs) {
        // fragments: A and B both contiguous k = 16*hi + s
        // (one contiguous 32B chunk each in m-/n-major LDS)
        v16h af[4], bf[4];
#pragma unroll
        for (int t = 0; t < 4; ++t) {
            int row = wm + t * 16 + ml;
            af[t] = *(const v16h*)&as[row * BKt + 16 * hi];
            int col = wn + t * 16 + ml;
            bf[t] = *(const v16h*)&bs[col * BKt + 16 * hi];
        }
#pragma unroll
        for (int i = 0; i < 4; ++i)
#pragma unroll
            for (int j = 0; j < 4; ++j)
                acc[i][j] = __builtin_amdgcn_wmma_f32_16x16x32_f16(
                    false, af[i], false, bf[j], (short)0, acc[i][j], false, false);
    };

    // ---- pipelined main loop, unrolled x2 (nk even, >= 2 for all shapes) ----
    const int nk = K / BKt;
    issue(0, 0);
    issue(1, BKt);
    for (int kt = 0; kt < nk; kt += 2) {
        // ---- half-step A: tile kt in buffer 0 (tile kt+1 stays in flight) ----
        WAIT_ASYNC6;
        __syncthreads();
        compute(As[0], Bs[0]);
        __syncthreads();                       // everyone done reading buf0
        const bool more0 = (kt + 2) < nk;
        if (more0) issue(0, (kt + 2) * BKt);

        // ---- half-step B: tile kt+1 in buffer 1 ----
        if (more0) { WAIT_ASYNC6; } else { WAIT_ASYNC0; }
        __syncthreads();
        compute(As[1], Bs[1]);
        __syncthreads();                       // everyone done reading buf1
        if ((kt + 3) < nk) issue(1, (kt + 3) * BKt);
    }

    // ---- epilogue: C/D layout — VGPR r, lane -> (M = r + 8*hi, N = ml) ----
    const float sc = sadd ? sadd[0] : 0.0f;
#pragma unroll
    for (int i = 0; i < 4; ++i) {
#pragma unroll
        for (int j = 0; j < 4; ++j) {
            int gn = n0 + wn + j * 16 + ml;
            float bv = (bias ? bias[gn] : 0.0f) + sc;
#pragma unroll
            for (int r = 0; r < 8; ++r) {
                int gm = m0 + wm + i * 16 + 8 * hi + r;
                float v = acc[i][j][r] + bv;
                if (act) v = tanhf(v);
                size_t off = (size_t)gm * N + gn;
                if (Cf) Cf[off] = v;
                if (Ch) Ch[off] = (_Float16)v;
            }
        }
    }
}

// =====================================================================
// Prep / reduction kernels
// =====================================================================
__global__ void zero_acc(float* p) { if (threadIdx.x < 8) p[threadIdx.x] = 0.0f; }

// fp32 [R,C] -> f16 transposed [C,R]  (LDS 32x32 tile transpose)
__global__ __launch_bounds__(256)
void convT_f16(const float* __restrict__ in, _Float16* __restrict__ out,
               int R, int C) {
    __shared__ float t[32][33];
    int tx = threadIdx.x & 31, ty = threadIdx.x >> 5;
    int c0 = blockIdx.x * 32, r0 = blockIdx.y * 32;
#pragma unroll
    for (int q = 0; q < 4; ++q) {
        int r = r0 + ty * 4 + q;
        t[ty * 4 + q][tx] = in[(size_t)r * C + c0 + tx];
    }
    __syncthreads();
#pragma unroll
    for (int q = 0; q < 4; ++q) {
        int c = c0 + ty * 4 + q;
        out[(size_t)c * R + r0 + tx] = (_Float16)t[tx][ty * 4 + q];
    }
}

__global__ __launch_bounds__(256)
void row_sums(const float* __restrict__ x, float* __restrict__ su) {
    __shared__ float red[256];
    int u = blockIdx.x, t = threadIdx.x;
    float s = 0.0f;
    for (int i = t; i < Ic; i += 256) s += fabsf(x[(size_t)u * Ic + i]);
    red[t] = s; __syncthreads();
    for (int st = 128; st > 0; st >>= 1) { if (t < st) red[t] += red[t + st]; __syncthreads(); }
    if (t == 0) su[u] = red[0];
}

__global__ __launch_bounds__(256)
void col_sums(const float* __restrict__ x, float* __restrict__ si) {
    int c = blockIdx.x * 256 + threadIdx.x;
    float s = 0.0f;
    for (int u = 0; u < Uc; ++u) s += fabsf(x[(size_t)u * Ic + c]);
    si[c] = s;
}

// user_ratings f16 [U,I] + item_ratings f16 [I,U] (LDS 32x32 transpose)
__global__ __launch_bounds__(256)
void norm_convert(const float* __restrict__ x, const float* __restrict__ su,
                  const float* __restrict__ si, _Float16* __restrict__ ur,
                  _Float16* __restrict__ ir) {
    __shared__ float tile[32][33];
    int tx = threadIdx.x & 31, ty = threadIdx.x >> 5;
    int i0 = blockIdx.x * 32, u0 = blockIdx.y * 32;
#pragma unroll
    for (int r = 0; r < 4; ++r) {
        int row = ty * 4 + r;
        int u = u0 + row, i = i0 + tx;
        float v = x[(size_t)u * Ic + i];
        ur[(size_t)u * Ic + i] = (_Float16)(v / fmaxf(su[u], 1e-12f));
        tile[row][tx] = v;
    }
    __syncthreads();
#pragma unroll
    for (int r = 0; r < 4; ++r) {
        int row = ty * 4 + r;
        int i = i0 + row, u = u0 + tx;
        float v = tile[tx][row];
        ir[(size_t)i * Uc + u] = (_Float16)(v / fmaxf(si[i], 1e-12f));
    }
}

// -Σ log_softmax(logits)*ratings per row, accumulated into acc (user tower)
__global__ __launch_bounds__(256)
void user_recon(const float* __restrict__ logits, const float* __restrict__ x,
                const float* __restrict__ su, float* __restrict__ acc) {
    __shared__ float red[256];
    int u = blockIdx.x, t = threadIdx.x;
    const float* Lg = logits + (size_t)u * Ic;
    float mx = -3.4e38f;
    for (int i = t; i < Ic; i += 256) mx = fmaxf(mx, Lg[i]);
    red[t] = mx; __syncthreads();
    for (int st = 128; st > 0; st >>= 1) { if (t < st) red[t] = fmaxf(red[t], red[t + st]); __syncthreads(); }
    mx = red[0]; __syncthreads();
    float se = 0.0f;
    for (int i = t; i < Ic; i += 256) se += expf(Lg[i] - mx);
    red[t] = se; __syncthreads();
    for (int st = 128; st > 0; st >>= 1) { if (t < st) red[t] += red[t + st]; __syncthreads(); }
    float lse = mx + logf(red[0]); __syncthreads();
    float inv = 1.0f / fmaxf(su[u], 1e-12f);
    float dot = 0.0f;
    for (int i = t; i < Ic; i += 256) dot += (Lg[i] - lse) * (x[(size_t)u * Ic + i] * inv);
    red[t] = dot; __syncthreads();
    for (int st = 128; st > 0; st >>= 1) { if (t < st) red[t] += red[t + st]; __syncthreads(); }
    if (t == 0) atomicAdd(acc, -red[0]);
}

// item tower: row length U, ratings taken from f16 item_ratings
__global__ __launch_bounds__(256)
void item_recon(const float* __restrict__ logits, const _Float16* __restrict__ ir,
                float* __restrict__ acc) {
    __shared__ float red[256];
    int i = blockIdx.x, t = threadIdx.x;
    const float* Lg = logits + (size_t)i * Uc;
    float mx = -3.4e38f;
    for (int u = t; u < Uc; u += 256) mx = fmaxf(mx, Lg[u]);
    red[t] = mx; __syncthreads();
    for (int st = 128; st > 0; st >>= 1) { if (t < st) red[t] = fmaxf(red[t], red[t + st]); __syncthreads(); }
    mx = red[0]; __syncthreads();
    float se = 0.0f;
    for (int u = t; u < Uc; u += 256) se += expf(Lg[u] - mx);
    red[t] = se; __syncthreads();
    for (int st = 128; st > 0; st >>= 1) { if (t < st) red[t] += red[t + st]; __syncthreads(); }
    float lse = mx + logf(red[0]); __syncthreads();
    float dot = 0.0f;
    for (int u = t; u < Uc; u += 256) dot += (Lg[u] - lse) * (float)ir[(size_t)i * Uc + u];
    red[t] = dot; __syncthreads();
    for (int st = 128; st > 0; st >>= 1) { if (t < st) red[t] += red[t + st]; __syncthreads(); }
    if (t == 0) atomicAdd(acc, -red[0]);
}

__global__ __launch_bounds__(256)
void kl_reduce(const float* __restrict__ mu, const float* __restrict__ lv, int n,
               float* __restrict__ acc) {
    __shared__ float red[256];
    int t = threadIdx.x; float s = 0.0f;
    for (size_t i = (size_t)blockIdx.x * 256 + t; i < (size_t)n; i += (size_t)gridDim.x * 256) {
        float m = mu[i], l = lv[i];
        s += expf(l) + m * m - l - 1.0f;
    }
    red[t] = s; __syncthreads();
    for (int st = 128; st > 0; st >>= 1) { if (t < st) red[t] += red[t + st]; __syncthreads(); }
    if (t == 0) atomicAdd(acc, red[0]);
}

__global__ __launch_bounds__(256)
void mse_reduce(const float* __restrict__ a, const float* __restrict__ b, int n,
                float* __restrict__ acc) {
    __shared__ float red[256];
    int t = threadIdx.x; float s = 0.0f;
    for (size_t i = (size_t)blockIdx.x * 256 + t; i < (size_t)n; i += (size_t)gridDim.x * 256) {
        float d = a[i] - b[i];
        s += d * d;
    }
    red[t] = s; __syncthreads();
    for (int st = 128; st > 0; st >>= 1) { if (t < st) red[t] += red[t + st]; __syncthreads(); }
    if (t == 0) atomicAdd(acc, red[0]);
}

__global__ void finalize_loss(const float* __restrict__ acc, float* __restrict__ out) {
    if (threadIdx.x == 0) {
        float total = acc[0] / (float)Uc
                    + acc[1] / (float)Ic
                    + 0.5f * acc[2] / (float)Uc
                    + 0.5f * acc[3] / (float)Ic
                    + 0.1f * (acc[4] / (float)((size_t)Ic * Lc)
                            + acc[5] / (float)((size_t)Uc * Lc));
        out[0] = total;
    }
}

// =====================================================================
// Host-side orchestration
// =====================================================================
extern "C" void kernel_launch(void* const* d_in, const int* in_sizes, int n_in,
                              void* d_out, int out_size, void* d_ws, size_t ws_size,
                              hipStream_t stream) {
    (void)in_sizes; (void)n_in; (void)out_size; (void)ws_size;

    // ---- inputs (setup_inputs order) ----
    const float* x      = (const float*)d_in[0];
    const float* ue_w1  = (const float*)d_in[1];  const float* ue_b1 = (const float*)d_in[2];
    const float* ue_w2  = (const float*)d_in[3];  const float* ue_b2 = (const float*)d_in[4];
    const float* ue_muw = (const float*)d_in[5];  const float* ue_mub= (const float*)d_in[6];
    const float* ue_lvw = (const float*)d_in[7];  const float* ue_lvb= (const float*)d_in[8];
    const float* ud_w1  = (const float*)d_in[9];  const float* ud_b1 = (const float*)d_in[10];
    const float* ud_w2  = (const float*)d_in[11]; const float* ud_b2 = (const float*)d_in[12];
    const float* ud_w3  = (const float*)d_in[13]; const float* ud_b3 = (const float*)d_in[14];
    const float* ie_w1  = (const float*)d_in[15]; const float* ie_b1 = (const float*)d_in[16];
    const float* ie_w2  = (const float*)d_in[17]; const float* ie_b2 = (const float*)d_in[18];
    const float* ie_muw = (const float*)d_in[19]; const float* ie_mub= (const float*)d_in[20];
    const float* ie_lvw = (const float*)d_in[21]; const float* ie_lvb= (const float*)d_in[22];
    const float* id_w1  = (const float*)d_in[23]; const float* id_b1 = (const float*)d_in[24];
    const float* id_w2  = (const float*)d_in[25]; const float* id_b2 = (const float*)d_in[26];
    const float* id_w3  = (const float*)d_in[27]; const float* id_b3 = (const float*)d_in[28];
    const float* u2i_w  = (const float*)d_in[29]; const float* u2i_b = (const float*)d_in[30];
    const float* i2u_w  = (const float*)d_in[31]; const float* i2u_b = (const float*)d_in[32];
    const float* bil_w  = (const float*)d_in[33]; const float* bil_b = (const float*)d_in[34];

    // ---- outputs (concatenated in return order) ----
    float* out   = (float*)d_out;
    float* pred  = out;                                    // [U,I]
    float* ulog  = out + (size_t)Uc * Ic;                  // [U,I]
    float* ilog  = out + 2 * (size_t)Uc * Ic;              // [I,U]
    float* uz    = out + 3 * (size_t)Uc * Ic;              // [U,L]
    float* iz    = uz + (size_t)Uc * Lc;                   // [I,L]
    float* tloss = iz + (size_t)Ic * Lc;                   // [1]

    // ---- workspace bump allocator (deterministic each call) ----
    char* wsp = (char*)d_ws;
    auto alloc = [&](size_t bytes) -> void* {
        void* p = (void*)wsp;
        wsp += (bytes + 255) & ~(size_t)255;
        return p;
    };
    float*     acc   = (float*)alloc(8 * sizeof(float));
    float*     sumU  = (float*)alloc((size_t)Uc * 4);
    float*     sumI  = (float*)alloc((size_t)Ic * 4);
    _Float16*  urh   = (_Float16*)alloc((size_t)Uc * Ic * 2);
    _Float16*  irh   = (_Float16*)alloc((size_t)Ic * Uc * 2);
    // f16 transposed weights  wt[N,K] from w[K,N]
    _Float16*  ue_w1t  = (_Float16*)alloc((size_t)Ic * H1c * 2);
    _Float16*  ue_w2t  = (_Float16*)alloc((size_t)H1c * H2c * 2);
    _Float16*  ue_muwt = (_Float16*)alloc((size_t)H2c * Lc * 2);
    _Float16*  ue_lvwt = (_Float16*)alloc((size_t)H2c * Lc * 2);
    _Float16*  ud_w1t  = (_Float16*)alloc((size_t)Lc * H2c * 2);
    _Float16*  ud_w2t  = (_Float16*)alloc((size_t)H2c * H1c * 2);
    _Float16*  ud_w3t  = (_Float16*)alloc((size_t)H1c * Ic * 2);
    _Float16*  ie_w1t  = (_Float16*)alloc((size_t)Uc * H1c * 2);
    _Float16*  ie_w2t  = (_Float16*)alloc((size_t)H1c * H2c * 2);
    _Float16*  ie_muwt = (_Float16*)alloc((size_t)H2c * Lc * 2);
    _Float16*  ie_lvwt = (_Float16*)alloc((size_t)H2c * Lc * 2);
    _Float16*  id_w1t  = (_Float16*)alloc((size_t)Lc * H2c * 2);
    _Float16*  id_w2t  = (_Float16*)alloc((size_t)H2c * H1c * 2);
    _Float16*  id_w3t  = (_Float16*)alloc((size_t)H1c * Uc * 2);
    _Float16*  u2i_wt  = (_Float16*)alloc((size_t)Lc * Lc * 2);
    _Float16*  i2u_wt  = (_Float16*)alloc((size_t)Lc * Lc * 2);
    _Float16*  bil_wt  = (_Float16*)alloc((size_t)Lc * Lc * 2);
    // f16 activations
    _Float16*  h1u = (_Float16*)alloc((size_t)Uc * H1c * 2);
    _Float16*  h2u = (_Float16*)alloc((size_t)Uc * H2c * 2);
    _Float16*  zuh = (_Float16*)alloc((size_t)Uc * Lc * 2);
    _Float16*  d1u = (_Float16*)alloc((size_t)Uc * H2c * 2);
    _Float16*  d2u = (_Float16*)alloc((size_t)Uc * H1c * 2);
    _Float16*  h1i = (_Float16*)alloc((size_t)Ic * H1c * 2);
    _Float16*  h2i = (_Float16*)alloc((size_t)Ic * H2c * 2);
    _Float16*  zih = (_Float16*)alloc((size_t)Ic * Lc * 2);
    _Float16*  d1i = (_Float16*)alloc((size_t)Ic * H2c * 2);
    _Float16*  d2i = (_Float16*)alloc((size_t)Ic * H1c * 2);
    _Float16*  ph  = (_Float16*)alloc((size_t)Uc * Lc * 2);
    // fp32 scratch
    float* lvu = (float*)alloc((size_t)Uc * Lc * 4);
    float* lvi = (float*)alloc((size_t)Ic * Lc * 4);
    float* ufi = (float*)alloc((size_t)Ic * Lc * 4);
    float* ifu = (float*)alloc((size_t)Uc * Lc * 4);

    // convert+transpose: w[R,C] fp32 -> wt[C,R] f16
    auto convT = [&](const float* srcW, _Float16* dstT, int R, int C) {
        convT_f16<<<dim3(C / 32, R / 32), 256, 0, stream>>>(srcW, dstT, R, C);
    };
    auto gemm = [&](const _Float16* A, const _Float16* B, const float* bias,
                    const float* sadd, float* Cf, _Float16* Ch,
                    int M, int N, int K, int act) {
        dim3 g((unsigned)(N / BNt), (unsigned)(M / BMt));
        gemm_f16<<<g, 256, 0, stream>>>(A, B, bias, sadd, Cf, Ch, M, N, K, act);
    };

    // ---- 0: loss accumulators ----
    zero_acc<<<1, 32, 0, stream>>>(acc);

    // ---- 1: L1 norms + f16 normalized matrices (incl. transpose) ----
    row_sums<<<Uc, 256, 0, stream>>>(x, sumU);
    col_sums<<<Ic / 256, 256, 0, stream>>>(x, sumI);
    norm_convert<<<dim3(Ic / 32, Uc / 32), 256, 0, stream>>>(x, sumU, sumI, urh, irh);

    // ---- 2: weights -> transposed f16 ----
    convT(ue_w1, ue_w1t, Ic, H1c);   convT(ue_w2, ue_w2t, H1c, H2c);
    convT(ue_muw, ue_muwt, H2c, Lc); convT(ue_lvw, ue_lvwt, H2c, Lc);
    convT(ud_w1, ud_w1t, Lc, H2c);   convT(ud_w2, ud_w2t, H2c, H1c);
    convT(ud_w3, ud_w3t, H1c, Ic);
    convT(ie_w1, ie_w1t, Uc, H1c);   convT(ie_w2, ie_w2t, H1c, H2c);
    convT(ie_muw, ie_muwt, H2c, Lc); convT(ie_lvw, ie_lvwt, H2c, Lc);
    convT(id_w1, id_w1t, Lc, H2c);   convT(id_w2, id_w2t, H2c, H1c);
    convT(id_w3, id_w3t, H1c, Uc);
    convT(u2i_w, u2i_wt, Lc, Lc);    convT(i2u_w, i2u_wt, Lc, Lc);
    convT(bil_w, bil_wt, Lc, Lc);

    // ---- 3: user tower ----
    gemm(urh, ue_w1t, ue_b1, nullptr, nullptr, h1u, Uc, H1c, Ic, 1);   // tanh
    gemm(h1u, ue_w2t, ue_b2, nullptr, nullptr, h2u, Uc, H2c, H1c, 1);  // tanh
    gemm(h2u, ue_muwt, ue_mub, nullptr, uz,   zuh, Uc, Lc, H2c, 0);    // user_z
    gemm(h2u, ue_lvwt, ue_lvb, nullptr, lvu, nullptr, Uc, Lc, H2c, 0); // logvar
    gemm(zuh, ud_w1t, ud_b1, nullptr, nullptr, d1u, Uc, H2c, Lc, 1);
    gemm(d1u, ud_w2t, ud_b2, nullptr, nullptr, d2u, Uc, H1c, H2c, 1);
    gemm(d2u, ud_w3t, ud_b3, nullptr, ulog, nullptr, Uc, Ic, H1c, 0);  // user_logits

    // ---- 4: item tower ----
    gemm(irh, ie_w1t, ie_b1, nullptr, nullptr, h1i, Ic, H1c, Uc, 1);
    gemm(h1i, ie_w2t, ie_b2, nullptr, nullptr, h2i, Ic, H2c, H1c, 1);
    gemm(h2i, ie_muwt, ie_mub, nullptr, iz,   zih, Ic, Lc, H2c, 0);    // item_z
    gemm(h2i, ie_lvwt, ie_lvb, nullptr, lvi, nullptr, Ic, Lc, H2c, 0);
    gemm(zih, id_w1t, id_b1, nullptr, nullptr, d1i, Ic, H2c, Lc, 1);
    gemm(d1i, id_w2t, id_b2, nullptr, nullptr, d2i, Ic, H1c, H2c, 1);
    gemm(d2i, id_w3t, id_b3, nullptr, ilog, nullptr, Ic, Uc, H1c, 0);  // item_logits

    // ---- 5: bilinear predictions: (uz @ bil_w) @ iz^T + bil_b ----
    gemm(zuh, bil_wt, nullptr, nullptr, nullptr, ph, Uc, Lc, Lc, 0);
    gemm(ph, zih, nullptr, bil_b, pred, nullptr, Uc, Ic, Lc, 0);       // zih IS [N,K]

    // ---- 6: cross projections ----
    gemm(zih, i2u_wt, i2u_b, nullptr, ufi, nullptr, Ic, Lc, Lc, 0);
    gemm(zuh, u2i_wt, u2i_b, nullptr, ifu, nullptr, Uc, Lc, Lc, 0);

    // ---- 7: losses ----
    user_recon<<<Uc, 256, 0, stream>>>(ulog, x, sumU, &acc[0]);
    item_recon<<<Ic, 256, 0, stream>>>(ilog, irh, &acc[1]);
    kl_reduce<<<512, 256, 0, stream>>>(uz, lvu, Uc * Lc, &acc[2]);
    kl_reduce<<<512, 256, 0, stream>>>(iz, lvi, Ic * Lc, &acc[3]);
    mse_reduce<<<512, 256, 0, stream>>>(ufi, iz, Ic * Lc, &acc[4]);
    mse_reduce<<<512, 256, 0, stream>>>(ifu, uz, Uc * Lc, &acc[5]);
    finalize_loss<<<1, 32, 0, stream>>>(acc, tloss);
}
